// DC_86543591015280
// MI455X (gfx1250) — matrix operand order repaired and checked by reading, not compiled, not run on hardware
//
#include <hip/hip_runtime.h>
#include <math.h>

// ---------------------------------------------------------------------------
// DCNv2 on gfx1250: all heavy math via v_wmma_f32_16x16x32_bf16 (wave32).
// A-operands staged in LDS in exact WMMA fragment order -> 32B ds loads;
// B-operands prepacked in fragment order -> coalesced 32B global loads.
// ---------------------------------------------------------------------------

typedef __bf16 bf16_t;
typedef bf16_t v16bf __attribute__((ext_vector_type(16)));
typedef bf16_t v8bf  __attribute__((ext_vector_type(8)));
typedef float  v8f   __attribute__((ext_vector_type(8)));

#define B_N   2
#define CI    128
#define CO    128
#define H_N   128
#define W_N   128
#define TW    16                 // pixels per M tile
#define NWT   (W_N / TW)         // 8 tiles per row
#define KTOT  (CI * 9)           // 1152
#define NCHK  (KTOT / 32)        // 36 K-chunks of 32
#define OMC   28                 // padded per-pixel offset/mask record

// workspace layout (bytes)
#define WQCONV_ELEMS ((CO / 16) * NCHK * 32 * 16)   // 147456 bf16
#define WQOM_ELEMS   (2 * NCHK * 32 * 16)           //  36864 bf16
#define WQCONV_BYTES (WQCONV_ELEMS * 2)
#define WQOM_BYTES   (WQOM_ELEMS * 2)
#define OFFMASK_OFF  (WQCONV_BYTES + WQOM_BYTES)

static __device__ __forceinline__ v8f wmma_bf16(v16bf a, v16bf b, v8f c) {
  // (neg_a, A, neg_b, B, c_mod, C, reuse_a, reuse_b)
  return __builtin_amdgcn_wmma_f32_16x16x32_bf16(false, a, false, b,
                                                 (short)0, c, false, false);
}

// fast sigmoid: v_exp_f32 + v_rcp_f32 (mask is in [0,1]; 1-ulp class is fine)
static __device__ __forceinline__ float fast_sigmoid(float x) {
  float e = __builtin_amdgcn_exp2f(x * -1.44269504f);
  return __builtin_amdgcn_rcpf(1.0f + e);
}

// within-32-group channel permutation so one lane's A fragment is contiguous:
// lo lanes (K 0..7,16..23) -> cp 0..15 ; hi lanes (K 8..15,24..31) -> cp 16..31
static __device__ __forceinline__ int cperm(int c) {
  return (c & ~31) | (c & 7) | ((c & 16) >> 1) | ((c & 8) << 1);
}

// ---------------------------------------------------------------------------
// Kernel 0: repack fp32 weights into bf16 B-fragment lane order.
// B (32x16): lane l holds column n = l&15; K(E) = (E&7) + 8*(l>=16) + 16*(E>>3).
// Linear order ((ntile*NCHK + chunk)*32 + lane)*16 + E -> one 32B load/lane.
// K ordering is k = tap*128 + c (tap-major, matching the A staging).
// ---------------------------------------------------------------------------
__global__ void dcn_prep_w(const float* __restrict__ w_p,
                           const float* __restrict__ w_m,
                           const float* __restrict__ w_conv,
                           bf16_t* __restrict__ wq_conv,
                           bf16_t* __restrict__ wq_om) {
  int i = blockIdx.x * blockDim.x + threadIdx.x;
  if (i < WQCONV_ELEMS) {
    int E = i & 15, lane = (i >> 4) & 31, g = i >> 9;
    int ntile = g / NCHK, chunk = g % NCHK;
    int n = ntile * 16 + (lane & 15);
    int k = chunk * 32 + (E & 7) + ((lane >> 4) << 3) + ((E >> 3) << 4);
    int tap = k / CI, c = k % CI;
    wq_conv[i] = (bf16_t)w_conv[(n * CI + c) * 9 + tap];
  } else if (i < WQCONV_ELEMS + WQOM_ELEMS) {
    int j = i - WQCONV_ELEMS;
    int E = j & 15, lane = (j >> 4) & 31, g = j >> 9;
    int ntile = g / NCHK, chunk = g % NCHK;
    int n = ntile * 16 + (lane & 15);
    int k = chunk * 32 + (E & 7) + ((lane >> 4) << 3) + ((E >> 3) << 4);
    int tap = k / CI, c = k % CI;
    float v = 0.f;
    if (n < 18)      v = w_p[(n * CI + c) * 9 + tap];        // offset weights
    else if (n < 27) v = w_m[((n - 18) * CI + c) * 9 + tap]; // mask weights
    wq_om[j] = (bf16_t)v;
  }
}

// ---------------------------------------------------------------------------
// Kernel 1: fused offset(18ch)+mask(9ch, sigmoid) 3x3 conv as implicit GEMM.
// One workgroup = (b, h, 16-pixel W tile). 2 waves = 2 N-tiles of 16 out ch.
// Patch staged channel-minor + bit-permuted -> single 32B LDS read per WMMA.
// ---------------------------------------------------------------------------
__global__ __launch_bounds__(64) void dcn_offmask(
    const float* __restrict__ x, const bf16_t* __restrict__ wq_om,
    const float* __restrict__ b_p, const float* __restrict__ b_m,
    float* __restrict__ offmask) {
  __shared__ bf16_t patch[3 * 18 * CI];   // [(r*18+col)*128 + cperm(c)]

  int wg = blockIdx.x;
  int wt = wg % NWT;
  int h  = (wg / NWT) % H_N;
  int b  = wg / (NWT * H_N);
  int tid = (int)threadIdx.x;

  // Stage 3x18 halo (zero-padded) for all 128 channels, fp32 -> bf16.
  for (int i = tid; i < CI * 3 * 18; i += 64) {
    int c = i / 54, rem = i % 54, r = rem / 18, col = rem % 18;
    int gr = h - 1 + r, gc = wt * TW - 1 + col;
    float v = 0.f;
    if (gr >= 0 && gr < H_N && gc >= 0 && gc < W_N)
      v = x[((b * CI + c) * H_N + gr) * W_N + gc];
    patch[(r * 18 + col) * CI + cperm(c)] = (bf16_t)v;
  }
  __syncthreads();

  int wave   = tid >> 5;            // N-tile id (0: out ch 0-15, 1: 16-31)
  int lane   = tid & 31;
  int m      = lane & 15;           // A-frag row (pixel within tile)
  int lhalf  = (lane >> 4) << 4;    // +16 permuted-channel offset, hi lanes
  int khalf8 = (lane >> 4) << 3;    // +8 M offset for D rows

  v8f acc = {};
  for (int chunk = 0; chunk < NCHK; ++chunk) {
    int tap = chunk >> 2, c0 = (chunk & 3) * 32;
    int ky = tap / 3, kx = tap % 3;
    // one contiguous 32B fragment read per lane
    v16bf a = *(const v16bf*)(&patch[(ky * 18 + (m + kx)) * CI + c0 + lhalf]);
    int nc = (chunk + 1 < NCHK) ? (chunk + 1) : chunk;
    __builtin_prefetch(wq_om + (((long)wave * NCHK + nc) * 32 + lane) * 16,
                       0, 3);
    v16bf bq = *(const v16bf*)(wq_om +
        (((long)wave * NCHK + chunk) * 32 + lane) * 16);
    acc = wmma_bf16(a, bq, acc);
  }

  // D layout: lane l -> N = l&15; VGPR v -> M = v + 8*(l>=16)
#pragma unroll
  for (int v = 0; v < 8; ++v) {
    int mp = v + khalf8;
    int n  = wave * 16 + (lane & 15);
    if (n < 27) {
      float val = acc[v] + (n < 18 ? b_p[n] : b_m[n - 18]);
      if (n >= 18) val = fast_sigmoid(val);   // mask channels
      offmask[(((b * H_N) + h) * W_N + (wt * TW + mp)) * OMC + n] = val;
    }
  }
}

// ---------------------------------------------------------------------------
// Kernel 2: modulated bilinear sampling -> LDS A in WMMA fragment order,
// then the final conv as M=16, N=128, K=1152 bf16 WMMA GEMM.
// Each wave owns 2 N-tiles and shares one A fragment per chunk between them.
// ---------------------------------------------------------------------------
__global__ __launch_bounds__(128) void dcn_sample_gemm(
    const float* __restrict__ x, const bf16_t* __restrict__ wq_conv,
    const float* __restrict__ offmask, float* __restrict__ out) {
  __shared__ bf16_t Afrag[NCHK * 32 * 16];   // fragment order, 36 KB

  int wg = blockIdx.x;
  int wt = wg % NWT;
  int h  = (wg / NWT) % H_N;
  int b  = wg / (NWT * H_N);
  int tid = (int)threadIdx.x;

  // ---- phase 1: deformable bilinear gather + mask modulation ----
  {
    int m  = tid & 15;          // pixel within tile (A-frag row)
    int cg = tid >> 4;          // 8 groups of 16 channels
    int w_pix = wt * TW + m;
    int pixbase = (((b * H_N) + h) * W_N + w_pix) * OMC;
    for (int tap = 0; tap < 9; ++tap) {
      float ox = offmask[pixbase + tap];           // row offset
      float oy = offmask[pixbase + 9 + tap];       // col offset
      float mk = offmask[pixbase + 18 + tap];      // modulation mask
      float px = ox + (float)(h + 1) + (float)(tap / 3 - 1);
      float py = oy + (float)(w_pix + 1) + (float)(tap % 3 - 1);
      float fx = floorf(px), fy = floorf(py);
      int qltx = min(max((int)fx, 0), H_N + 1);
      int qlty = min(max((int)fy, 0), W_N + 1);
      int qrbx = min(max((int)fx + 1, 0), H_N + 1);
      int qrby = min(max((int)fy + 1, 0), W_N + 1);
      float pxc = fminf(fmaxf(px, 0.f), (float)(H_N + 1));
      float pyc = fminf(fmaxf(py, 0.f), (float)(W_N + 1));
      float glt = (1.f + ((float)qltx - pxc)) * (1.f + ((float)qlty - pyc));
      float grb = (1.f - ((float)qrbx - pxc)) * (1.f - ((float)qrby - pyc));
      float glb = (1.f + ((float)qltx - pxc)) * (1.f - ((float)qrby - pyc));
      float grt = (1.f - ((float)qrbx - pxc)) * (1.f + ((float)qlty - pyc));
      bool vlt = (qltx >= 1 && qltx <= H_N && qlty >= 1 && qlty <= W_N);
      bool vrb = (qrbx >= 1 && qrbx <= H_N && qrby >= 1 && qrby <= W_N);
      bool vlb = (qltx >= 1 && qltx <= H_N && qrby >= 1 && qrby <= W_N);
      bool vrt = (qrbx >= 1 && qrbx <= H_N && qlty >= 1 && qlty <= W_N);
      int ilt = (qltx - 1) * W_N + (qlty - 1);
      int irb = (qrbx - 1) * W_N + (qrby - 1);
      int ilb = (qltx - 1) * W_N + (qrby - 1);
      int irt = (qrbx - 1) * W_N + (qlty - 1);
      v8bf lo = {}, hi = {};   // cc 0..7 -> lane m ; cc 8..15 -> lane m+16
#pragma unroll
      for (int cc = 0; cc < 16; ++cc) {
        int c = cg * 16 + cc;
        const float* xc = x + ((long)(b * CI + c)) * (H_N * W_N);
        float val = glt * (vlt ? xc[ilt] : 0.f)
                  + grb * (vrb ? xc[irb] : 0.f)
                  + glb * (vlb ? xc[ilb] : 0.f)
                  + grt * (vrt ? xc[irt] : 0.f);
        val *= mk;
        bf16_t bv = (bf16_t)val;
        if (cc < 8) lo[cc] = bv; else hi[cc - 8] = bv;
      }
      // fragment-order addresses: two 16B LDS stores per (tap, cgroup)
      int chunk = tap * 4 + (cg >> 1);
      int eh = (cg & 1) * 8;           // E offset 0 or 8
      *(v8bf*)(&Afrag[(chunk * 32 + m) * 16 + eh])       = lo;
      *(v8bf*)(&Afrag[(chunk * 32 + m + 16) * 16 + eh])  = hi;
    }
  }
  __syncthreads();

  // ---- phase 2: WMMA GEMM, 4 waves x (2 N-tiles each) x 36 K-chunks.
  //      One A fragment read feeds two WMMAs (dual accumulators). ----
  int wave   = tid >> 5;
  int lane   = tid & 31;
  int khalf8 = (lane >> 4) << 3;
  int nt0 = wave * 2, nt1 = wave * 2 + 1;
  v8f acc0 = {}, acc1 = {};
  for (int chunk = 0; chunk < NCHK; ++chunk) {
    // one contiguous 32B A read per lane, shared by both N-tiles
    v16bf a = *(const v16bf*)(&Afrag[(chunk * 32 + lane) * 16]);
    int nc = (chunk + 1 < NCHK) ? (chunk + 1) : chunk;
    __builtin_prefetch(
        wq_conv + (((long)nt0 * NCHK + nc) * 32 + lane) * 16, 0, 3);
    __builtin_prefetch(
        wq_conv + (((long)nt1 * NCHK + nc) * 32 + lane) * 16, 0, 3);
    v16bf b0 = *(const v16bf*)(wq_conv +
        (((long)nt0 * NCHK + chunk) * 32 + lane) * 16);
    v16bf b1 = *(const v16bf*)(wq_conv +
        (((long)nt1 * NCHK + chunk) * 32 + lane) * 16);
    acc0 = wmma_bf16(a, b0, acc0);
    acc1 = wmma_bf16(a, b1, acc1);
  }
#pragma unroll
  for (int v = 0; v < 8; ++v) {
    int mp = v + khalf8;
    int n0 = nt0 * 16 + (lane & 15);
    int n1 = nt1 * 16 + (lane & 15);
    out[((b * CO + n0) * H_N + h) * W_N + (wt * TW + mp)] = acc0[v];
    out[((b * CO + n1) * H_N + h) * W_N + (wt * TW + mp)] = acc1[v];
  }
}

// ---------------------------------------------------------------------------
extern "C" void kernel_launch(void* const* d_in, const int* in_sizes, int n_in,
                              void* d_out, int out_size, void* d_ws, size_t ws_size,
                              hipStream_t stream) {
  const float* x      = (const float*)d_in[0];
  const float* w_p    = (const float*)d_in[1];
  const float* b_p    = (const float*)d_in[2];
  const float* w_m    = (const float*)d_in[3];
  const float* b_m    = (const float*)d_in[4];
  const float* w_conv = (const float*)d_in[5];
  float* out = (float*)d_out;

  char* ws = (char*)d_ws;
  bf16_t* wq_conv = (bf16_t*)(ws);
  bf16_t* wq_om   = (bf16_t*)(ws + WQCONV_BYTES);
  float*  offmask = (float*)(ws + OFFMASK_OFF);
  (void)in_sizes; (void)n_in; (void)out_size; (void)ws_size;

  int prep_total = WQCONV_ELEMS + WQOM_ELEMS;
  dcn_prep_w<<<(prep_total + 255) / 256, 256, 0, stream>>>(
      w_p, w_m, w_conv, wq_conv, wq_om);

  int ntiles = B_N * H_N * NWT;   // 2048 workgroups
  dcn_offmask<<<ntiles, 64, 0, stream>>>(x, wq_om, b_p, b_m, offmask);
  dcn_sample_gemm<<<ntiles, 128, 0, stream>>>(x, wq_conv, offmask, out);
}
